// NNNet_59090160059187
// MI455X (gfx1250) — matrix operand order repaired and checked by reading, not compiled, not run on hardware
//
#include <hip/hip_runtime.h>

typedef __attribute__((ext_vector_type(16))) __bf16 v16bf;
typedef __attribute__((ext_vector_type(2)))  __bf16 v2bf;
typedef __attribute__((ext_vector_type(8)))  float  v8f;

#define H  32
#define FE 8
#define TT 10
#define EPSX 1e-5f

// ---- CDNA5 async global->LDS copy helpers (ASYNCcnt-tracked) ----------------
__device__ __forceinline__ void async_g2l_b128(void* lds, const void* gmem) {
  // Flat LDS addresses carry the LDS byte offset in the low 32 bits.
  unsigned loff = (unsigned)(unsigned long long)(uintptr_t)lds;
  unsigned long long ga = (unsigned long long)(uintptr_t)gmem;
  asm volatile("global_load_async_to_lds_b128 %0, %1, off"
               :: "v"(loff), "v"(ga) : "memory");
}
__device__ __forceinline__ void async_wait0() {
  asm volatile("s_wait_asynccnt 0x0" ::: "memory");
}

__global__ void k_zero(float* __restrict__ p, int n) {
  int i = blockIdx.x * 256 + threadIdx.x;
  if (i < n) p[i] = 0.f;
}

__global__ void k_deg(const int* __restrict__ batch, float* __restrict__ deg, int n) {
  int i = blockIdx.x * 256 + threadIdx.x;
  if (i < n) atomicAdd(&deg[batch[i]], 1.0f);
}

__global__ void k_scale(const int* __restrict__ batch, const float* __restrict__ deg,
                        float* __restrict__ scale, int n) {
  int i = blockIdx.x * 256 + threadIdx.x;
  if (i < n) {
    float d = deg[batch[i]];
    scale[i] = rsqrtf(d > 0.f ? d : 1.f);
  }
}

// Fused per-edge NNConv message GEMM:
//   Z[e, k]       = ea[e, k/IN] * feat[src[e], k%IN]    k <  FE*IN
//   Z[e, FE*IN+i] = feat[src[e], i]                     bias rows
//   msg[e,:] = Z[e,:] @ [nn_w; nn_b], scatter-add into agg[dst[e],:]
// One wave: 16 edges x 32 outputs (2 bf16 WMMA accumulators), 8 waves/block.
template<int IN, int KPAD>
__global__ __launch_bounds__(256)
void k_edge_msg(const float* __restrict__ feat,   // [N, IN]
                const float* __restrict__ eattr,  // [E, FE]
                const int*   __restrict__ eidx,   // [2, E]
                const float* __restrict__ nn_w,   // [FE*IN*H] flat
                const float* __restrict__ nn_b,   // [IN*H]
                float* __restrict__ agg,          // [N, H]
                int E)
{
  constexpr int KSTEPS = KPAD / 32;
  constexpr int KPROD  = FE * IN;
  constexpr int KTOT   = (FE + 1) * IN;
  constexpr int XS     = IN + 4;      // 16B-aligned rows, gcd(XS,64)=4 -> 16 banks
  constexpr int ESV    = 12;          // 48B rows for edge_attr
  constexpr int CH     = IN / 4;      // 16B chunks per feature row

  __shared__ __align__(16) float ea_s[128 * ESV];
  __shared__ __align__(16) float xs_s[128 * XS];
  __shared__ int  src_s[128];
  __shared__ int  dst_s[128];
  // bf16 weights laid out so one lane's whole B fragment is contiguous:
  //   wq[ ((ks*2 + half)*32 + n)*8 + r ] = { W[2*kp, n], W[2*kp+1, n] },
  //   kp = ks*16 + half*4 + (r>>2)*8 + (r&3)  -> per-lane read = one aligned 32B v16bf.
  __shared__ __align__(32) v2bf wq[(KPAD / 2) * H];

  const int tid   = threadIdx.x;
  const int ebase = blockIdx.x * 128;
  const bool full = (ebase + 128) <= E;

  if (tid < 128) {
    int e = ebase + tid;
    if (e < E) { src_s[tid] = eidx[e]; dst_s[tid] = eidx[E + e]; }
    else       { src_s[tid] = 0;       dst_s[tid] = -1; }
  }

  // Stage combined weight matrix [nn_w ; nn_b ; 0-pad] as bf16 (fragment order).
  for (int j = tid; j < (KPAD / 2) * H; j += 256) {
    int r      = j & 7;
    int row    = j >> 3;
    int n      = row & 31;
    int halfks = row >> 5;              // = ks*2 + half
    int kp     = (halfks >> 1) * 16 + (halfks & 1) * 4 + (r >> 2) * 8 + (r & 3);
    int f0     = (2 * kp) * H + n;
    int f1     = f0 + H;
    float w0 = (f0 < KPROD * H) ? nn_w[f0] : (f0 < KTOT * H ? nn_b[f0 - KPROD * H] : 0.f);
    float w1 = (f1 < KPROD * H) ? nn_w[f1] : (f1 < KTOT * H ? nn_b[f1 - KPROD * H] : 0.f);
    v2bf p; p.x = (__bf16)w0; p.y = (__bf16)w1;
    wq[j] = p;
  }

  // Edge attributes: async bulk copy (full blocks) / guarded scalar (tail).
  if (full) {
    int el = tid >> 1, c = tid & 1;
    async_g2l_b128(&ea_s[el * ESV + c * 4],
                   eattr + (size_t)(ebase + el) * FE + c * 4);
  } else {
    for (int j = tid; j < 128 * FE; j += 256) {
      int el = j >> 3, f = j & 7;
      int e = ebase + el;
      ea_s[el * ESV + f] = (e < E) ? eattr[(size_t)e * FE + f] : 0.f;
    }
  }
  __syncthreads();   // src_s visible for the gather below

  // Per-edge feature gather: per-lane async B128 (src is always a valid node).
  for (int j = tid; j < 128 * CH; j += 256) {
    int el = j / CH, c = j % CH;
    async_g2l_b128(&xs_s[el * XS + c * 4],
                   feat + (size_t)src_s[el] * IN + c * 4);
  }
  async_wait0();
  __syncthreads();

  const int lane = tid & 31;
  const int wave = tid >> 5;
  const int half = lane >> 4;
  const int nrow = lane & 15;
  const int eA   = wave * 16 + nrow;   // A-fragment edge row (M index)

  v8f acc0 = {};
  v8f acc1 = {};
  #pragma unroll
  for (int ks = 0; ks < KSTEPS; ++ks) {
    v16bf a;
    #pragma unroll
    for (int r = 0; r < 8; ++r) {
      // A 16x32 bf16 layout: lanes 0-15 K=0..7,16..23; lanes 16-31 K=8..15,24..31
      int k = ks * 32 + ((r < 4) ? 2 * r : 16 + 2 * (r - 4)) + half * 8;
      float av0, av1;
      if (k < KPROD) {
        int f = k / IN, i = k % IN;     // pair never straddles f (k even, IN even)
        float  ev = ea_s[eA * ESV + f];
        float2 xv = *(const float2*)&xs_s[eA * XS + i];
        av0 = ev * xv.x; av1 = ev * xv.y;
      } else if (k < KTOT) {
        float2 xv = *(const float2*)&xs_s[eA * XS + (k - KPROD)];
        av0 = xv.x; av1 = xv.y;
      } else {
        av0 = 0.f; av1 = 0.f;
      }
      a[2 * r]     = (__bf16)av0;
      a[2 * r + 1] = (__bf16)av1;
    }
    // B 32x16 bf16 fragments: contiguous 32B per lane (2x ds_load_b128).
    int row0 = (ks * 2 + half) * 32 + nrow;
    v16bf b0 = *(const v16bf*)&wq[row0 * 8];
    v16bf b1 = *(const v16bf*)&wq[(row0 + 16) * 8];
    acc0 = __builtin_amdgcn_wmma_f32_16x16x32_bf16(false, a, false, b0, (short)0, acc0, false, false);
    acc1 = __builtin_amdgcn_wmma_f32_16x16x32_bf16(false, a, false, b1, (short)0, acc1, false, false);
  }

  // C/D layout: lane(0-15) -> M=r, N=lane ; lane(16-31) -> M=8+r, N=lane-16
  #pragma unroll
  for (int r = 0; r < 8; ++r) {
    int el = wave * 16 + half * 8 + r;
    int d  = dst_s[el];
    if (d >= 0) {
      atomicAdd(&agg[d * H + nrow],      acc0[r]);
      atomicAdd(&agg[d * H + 16 + nrow], acc1[r]);
    }
  }
}

// out = relu(bn(graph_size_norm(agg + feat@root + bias))); re-zeros agg element.
template<int IN>
__global__ void k_node_bn(const float* __restrict__ feat,
                          const float* __restrict__ root,   // [IN, H]
                          const float* __restrict__ bias,
                          float* __restrict__ agg,
                          const float* __restrict__ scale,
                          const float* __restrict__ bg, const float* __restrict__ bb,
                          const float* __restrict__ bm, const float* __restrict__ bv,
                          float* __restrict__ out, int n)
{
  int idx = blockIdx.x * 256 + threadIdx.x;
  if (idx >= n * H) return;
  int node = idx >> 5, o = idx & 31;
  float val = agg[idx];
  agg[idx] = 0.f;                       // ready for next layer / next replay
  #pragma unroll
  for (int i = 0; i < IN; ++i) val += feat[node * IN + i] * root[i * H + o];
  val += bias[o];
  val *= scale[node];
  val = (val - bm[o]) * (bg[o] * rsqrtf(bv[o] + EPSX)) + bb[o];
  out[idx] = fmaxf(val, 0.f);
}

// layer-3 epilogue: out3 = agg + feat@root + bias, atomically pooled per graph.
__global__ void k_node_pool(const float* __restrict__ feat,
                            const float* __restrict__ root,  // [H, H]
                            const float* __restrict__ bias,
                            float* __restrict__ agg,
                            const int* __restrict__ batch,
                            float* __restrict__ pooled, int n)
{
  int idx = blockIdx.x * 256 + threadIdx.x;
  if (idx >= n * H) return;
  int node = idx >> 5, o = idx & 31;
  float val = agg[idx];
  agg[idx] = 0.f;
  #pragma unroll
  for (int i = 0; i < H; ++i) val += feat[node * H + i] * root[i * H + o];
  val += bias[o];
  atomicAdd(&pooled[batch[node] * H + o], val);
}

__global__ void k_head(const float* __restrict__ pooled, const float* __restrict__ deg,
                       const float* __restrict__ w1, const float* __restrict__ b1,
                       const float* __restrict__ w2, const float* __restrict__ b2,
                       float* __restrict__ out, int G)
{
  int g = blockIdx.x * 256 + threadIdx.x;
  if (g >= G) return;
  float inv = 1.f / fmaxf(deg[g], 1.f);
  float p[H], hid[H];
  #pragma unroll
  for (int i = 0; i < H; ++i) p[i] = pooled[g * H + i] * inv;
  #pragma unroll
  for (int j = 0; j < H; ++j) {
    float s = b1[j];
    #pragma unroll
    for (int i = 0; i < H; ++i) s += p[i] * w1[i * H + j];
    hid[j] = fmaxf(s, 0.f);
  }
  #pragma unroll
  for (int t = 0; t < TT; ++t) {
    float s = b2[t];
    #pragma unroll
    for (int j = 0; j < H; ++j) s += hid[j] * w2[j * TT + t];
    out[g * TT + t] = s;
  }
}

extern "C" void kernel_launch(void* const* d_in, const int* in_sizes, int n_in,
                              void* d_out, int out_size, void* d_ws, size_t ws_size,
                              hipStream_t stream)
{
  (void)n_in; (void)out_size; (void)ws_size;
  const float* x       = (const float*)d_in[0];
  const float* eattr   = (const float*)d_in[1];
  const int*   eidx    = (const int*)d_in[2];
  const int*   batch   = (const int*)d_in[3];
  const float* c1_nn_w = (const float*)d_in[4];
  const float* c1_nn_b = (const float*)d_in[5];
  const float* c1_root = (const float*)d_in[6];
  const float* c1_bias = (const float*)d_in[7];
  const float* c2_nn_w = (const float*)d_in[8];
  const float* c2_nn_b = (const float*)d_in[9];
  const float* c2_root = (const float*)d_in[10];
  const float* c2_bias = (const float*)d_in[11];
  const float* c3_nn_w = (const float*)d_in[12];
  const float* c3_nn_b = (const float*)d_in[13];
  const float* c3_root = (const float*)d_in[14];
  const float* c3_bias = (const float*)d_in[15];
  const float* bn1_g   = (const float*)d_in[16];
  const float* bn1_b   = (const float*)d_in[17];
  const float* bn1_m   = (const float*)d_in[18];
  const float* bn1_v   = (const float*)d_in[19];
  const float* bn2_g   = (const float*)d_in[20];
  const float* bn2_b   = (const float*)d_in[21];
  const float* bn2_m   = (const float*)d_in[22];
  const float* bn2_v   = (const float*)d_in[23];
  const float* mlp_w1  = (const float*)d_in[24];
  const float* mlp_b1  = (const float*)d_in[25];
  const float* mlp_w2  = (const float*)d_in[26];
  const float* mlp_b2  = (const float*)d_in[27];

  const int N = in_sizes[0] / 16;
  const int E = in_sizes[1] / FE;
  const int G = 256;

  float* ws     = (float*)d_ws;
  float* agg    = ws;                         // [N, H]
  float* h1     = agg   + (size_t)N * H;      // [N, H]
  float* h2     = h1    + (size_t)N * H;      // [N, H]
  float* scale  = h2    + (size_t)N * H;      // [N]
  float* deg    = scale + N;                  // [G]
  float* pooled = deg   + G;                  // [G, H]

  dim3 b256(256);
  k_zero<<<(N * H + 255) / 256, b256, 0, stream>>>(agg, N * H);
  k_zero<<<(G + 255) / 256,     b256, 0, stream>>>(deg, G);
  k_zero<<<(G * H + 255) / 256, b256, 0, stream>>>(pooled, G * H);
  k_deg  <<<(N + 255) / 256,    b256, 0, stream>>>(batch, deg, N);
  k_scale<<<(N + 255) / 256,    b256, 0, stream>>>(batch, deg, scale, N);

  const int egrid = (E + 127) / 128;
  const int ngrid = (N * H + 255) / 256;

  // Layer 1: IN=16, K = 8*16 + 16 = 144 -> pad 160
  k_edge_msg<16, 160><<<egrid, b256, 0, stream>>>(x, eattr, eidx, c1_nn_w, c1_nn_b, agg, E);
  k_node_bn<16><<<ngrid, b256, 0, stream>>>(x, c1_root, c1_bias, agg, scale,
                                            bn1_g, bn1_b, bn1_m, bn1_v, h1, N);
  // Layer 2: IN=32, K = 8*32 + 32 = 288
  k_edge_msg<32, 288><<<egrid, b256, 0, stream>>>(h1, eattr, eidx, c2_nn_w, c2_nn_b, agg, E);
  k_node_bn<32><<<ngrid, b256, 0, stream>>>(h1, c2_root, c2_bias, agg, scale,
                                            bn2_g, bn2_b, bn2_m, bn2_v, h2, N);
  // Layer 3: IN=32, no norm/bn; pooled accumulation
  k_edge_msg<32, 288><<<egrid, b256, 0, stream>>>(h2, eattr, eidx, c3_nn_w, c3_nn_b, agg, E);
  k_node_pool<<<ngrid, b256, 0, stream>>>(h2, c3_root, c3_bias, agg, batch, pooled, N);

  k_head<<<1, b256, 0, stream>>>(pooled, deg, mlp_w1, mlp_b1, mlp_w2, mlp_b2, (float*)d_out, G);
}
